// GraphGIN_49744311222604
// MI455X (gfx1250) — matrix operand order repaired and checked by reading, not compiled, not run on hardware
//
#include <hip/hip_runtime.h>
#include <hip/hip_bf16.h>

#define N_NODES  100000
#define N_EDGES  1600000
#define N_FEATS  128
#define HIDDEN   20
#define N_GRAPHS 256
#define N_CLASSES 2

typedef __attribute__((ext_vector_type(16))) _Float16 v16h;
typedef __attribute__((ext_vector_type(8)))  float    v8f;

// ---------------------------------------------------------------------------
// float4 copy (n4 = number of float4 elements)
__global__ void k_copy4(const float4* __restrict__ src, float4* __restrict__ dst, int n4) {
    int i = blockIdx.x * 256 + threadIdx.x;
    if (i < n4) dst[i] = src[i];
}

// ---------------------------------------------------------------------------
// zero pool scratch: mx (uint, ==0.0f), sums, counts
__global__ void k_init_pool(unsigned* mx, float* sm, float* cnt) {
    int tid = threadIdx.x;
    for (int i = tid; i < N_GRAPHS * HIDDEN; i += 256) { mx[i] = 0u; sm[i] = 0.0f; }
    for (int i = tid; i < N_GRAPHS; i += 256) cnt[i] = 0.0f;
}

// ---------------------------------------------------------------------------
// Convert one weight matrix W[Kreal x HIDDEN] (f32) into WMMA B-fragment order:
//   frag = kt*2 + nt  (kt = K/32 tile, nt = N/16 tile), 512 halves per frag:
//   Wfrag[frag*512 + lane*16 + i] = W[ kt*32 + (lane>>4)*8 + (i&7) + (i>>3)*16 ][ (lane&15) + nt*16 ]
// so each lane's v16h B fragment is 32 contiguous bytes (-> ds_load_b128 x2).
__device__ void convert_frag(const float* W, _Float16* Wfrag, int K, int Kreal, int tid) {
    int total = (K / 32) * 2 * 512;                // == K*32
    for (int idx = tid; idx < total; idx += 256) {
        int frag = idx >> 9;
        int r    = idx & 511;
        int lane = r >> 4;
        int i    = r & 15;
        int kt   = frag >> 1;
        int nt   = frag & 1;
        int n    = (lane & 15) + nt * 16;
        int k    = kt * 32 + (lane >> 4) * 8 + (i & 7) + (i >> 3) * 16;
        float v  = (k < Kreal && n < HIDDEN) ? W[k * HIDDEN + n] : 0.0f;
        Wfrag[idx] = (_Float16)v;
    }
}

__global__ void k_convert_w(const float* __restrict__ W1, const float* __restrict__ W2,
                            const float* __restrict__ W3,
                            const float* __restrict__ b1, const float* __restrict__ b2,
                            const float* __restrict__ b3,
                            _Float16* __restrict__ W1f, _Float16* __restrict__ W2f,
                            _Float16* __restrict__ W3f, float* __restrict__ bPad) {
    int tid = threadIdx.x;
    convert_frag(W1, W1f, N_FEATS, N_FEATS, tid);
    convert_frag(W2, W2f, 32, HIDDEN, tid);
    convert_frag(W3, W3f, 32, HIDDEN, tid);
    if (tid < 96) {
        int layer = tid >> 5, n = tid & 31;
        const float* b = (layer == 0) ? b1 : (layer == 1) ? b2 : b3;
        bPad[layer * 32 + n] = (n < HIDDEN) ? b[n] : 0.0f;  // padded cols: bias 0
    }
}

// ---------------------------------------------------------------------------
// layer-1 scatter: one wave32 per edge, float4 per lane (128 feats)
__global__ void k_scatter128(const float* __restrict__ x, const int* __restrict__ ei,
                             float* __restrict__ agg) {
    int t = blockIdx.x * 256 + threadIdx.x;
    int e = t >> 5, lane = t & 31;
    if (e >= N_EDGES) return;
    int s = ei[e];
    int d = ei[N_EDGES + e];
    float4 v = *(const float4*)(x + (size_t)s * N_FEATS + lane * 4);
    float* p = agg + (size_t)d * N_FEATS + lane * 4;
    atomicAdd(p + 0, v.x);
    atomicAdd(p + 1, v.y);
    atomicAdd(p + 2, v.z);
    atomicAdd(p + 3, v.w);
}

// layers 2/3 scatter: flat over (edge, feat) pairs -> all 32 lanes useful
__global__ void k_scatter20(const float* __restrict__ h, const int* __restrict__ ei,
                            float* __restrict__ agg) {
    int idx = blockIdx.x * 256 + threadIdx.x;
    if (idx >= N_EDGES * HIDDEN) return;
    int e = idx / HIDDEN;
    int f = idx - e * HIDDEN;
    int s = ei[e];
    int d = ei[N_EDGES + e];
    atomicAdd(agg + (size_t)d * 32 + f, h[(size_t)s * 32 + f]);
}

// ---------------------------------------------------------------------------
// load one 16x32 A tile (rows rowC..rowC+15 clamped) as a WMMA A fragment
__device__ __forceinline__ v16h load_a_frag(const float* __restrict__ A, int row,
                                            int K, int kk, int half) {
    const float* Arow = A + (size_t)row * K;
    int kb = kk + half * 8;
    float4 a0 = *(const float4*)(Arow + kb);
    float4 a1 = *(const float4*)(Arow + kb + 4);
    float4 a2 = *(const float4*)(Arow + kb + 16);
    float4 a3 = *(const float4*)(Arow + kb + 20);
    v16h av;
    av[0]  = (_Float16)a0.x; av[1]  = (_Float16)a0.y;
    av[2]  = (_Float16)a0.z; av[3]  = (_Float16)a0.w;
    av[4]  = (_Float16)a1.x; av[5]  = (_Float16)a1.y;
    av[6]  = (_Float16)a1.z; av[7]  = (_Float16)a1.w;
    av[8]  = (_Float16)a2.x; av[9]  = (_Float16)a2.y;
    av[10] = (_Float16)a2.z; av[11] = (_Float16)a2.w;
    av[12] = (_Float16)a3.x; av[13] = (_Float16)a3.y;
    av[14] = (_Float16)a3.z; av[15] = (_Float16)a3.w;
    return av;
}

// WMMA GEMM + bias + ReLU: Out[N x 32] = relu(A[N x K] @ W[K x 32] + bias)
// 256 threads = 8 waves; each wave computes a 32x32 tile (4 WMMAs / k-step,
// B fragments reused across both M tiles). Wfrag is in fragment order.
// If Out2 != nullptr the result is also written there (seeds next layer's agg;
// Out2 may alias A: each wave reads only its own rows, all reads complete
// before its stores issue, and row sets are disjoint across waves/blocks).
__global__ void k_gemm_relu(const float* __restrict__ A, const _Float16* __restrict__ Wfrag,
                            const float* __restrict__ biasPad, float* __restrict__ Out,
                            float* __restrict__ Out2, int K) {
    __shared__ _Float16 sW[N_FEATS * 32];   // max 8 KB (K<=128), fragment order
    int tid = threadIdx.x;
    for (int i = tid; i < K * 32; i += 256) sW[i] = Wfrag[i];
    __syncthreads();

    int wave = tid >> 5;
    int lane = tid & 31;
    int half = lane >> 4;
    int n0   = lane & 15;
    int rowBase = blockIdx.x * 256 + wave * 32;

    int rowL0 = rowBase + n0;
    int rowL1 = rowBase + 16 + n0;
    if (rowL0 > N_NODES - 1) rowL0 = N_NODES - 1;   // clamp loads; stores guarded
    if (rowL1 > N_NODES - 1) rowL1 = N_NODES - 1;

    v8f c00 = {}, c01 = {}, c10 = {}, c11 = {};

    for (int kk = 0; kk < K; kk += 32) {
        v16h av0 = load_a_frag(A, rowL0, K, kk, half);
        v16h av1 = load_a_frag(A, rowL1, K, kk, half);
        int kt = kk >> 5;
        v16h b0 = *(const v16h*)(sW + (kt * 2 + 0) * 512 + lane * 16);
        v16h b1 = *(const v16h*)(sW + (kt * 2 + 1) * 512 + lane * 16);
        c00 = __builtin_amdgcn_wmma_f32_16x16x32_f16(false, av0, false, b0,
                                                     (short)0, c00, false, false);
        c01 = __builtin_amdgcn_wmma_f32_16x16x32_f16(false, av0, false, b1,
                                                     (short)0, c01, false, false);
        c10 = __builtin_amdgcn_wmma_f32_16x16x32_f16(false, av1, false, b0,
                                                     (short)0, c10, false, false);
        c11 = __builtin_amdgcn_wmma_f32_16x16x32_f16(false, av1, false, b1,
                                                     (short)0, c11, false, false);
    }

    float bias0 = biasPad[n0];        // padded bias: cols 20..31 are 0
    float bias1 = biasPad[n0 + 16];

    // C layout: VGPR v -> row v (lanes 0-15) / row v+8 (lanes 16-31), col = n0 (+16)
    float r0v[8], r1v[8], r2v[8], r3v[8];
#pragma unroll
    for (int v = 0; v < 8; v++) {
        float a = c00[v] + bias0; r0v[v] = a < 0.0f ? 0.0f : a;
        float b = c01[v] + bias1; r1v[v] = b < 0.0f ? 0.0f : b;
        float c = c10[v] + bias0; r2v[v] = c < 0.0f ? 0.0f : c;
        float d = c11[v] + bias1; r3v[v] = d < 0.0f ? 0.0f : d;
    }

    int rowS0 = rowBase + half * 8;        // rows rowS0..rowS0+7 (tile 0)
    int rowS1 = rowS0 + 16;                // tile 1
    if (rowBase + 32 <= N_NODES) {         // wave-uniform fast path: no guards
#pragma unroll
        for (int v = 0; v < 8; v++) {
            size_t o0 = (size_t)(rowS0 + v) * 32 + n0;
            size_t o1 = (size_t)(rowS1 + v) * 32 + n0;
            Out[o0] = r0v[v];  Out[o0 + 16] = r1v[v];
            Out[o1] = r2v[v];  Out[o1 + 16] = r3v[v];
        }
        if (Out2) {
#pragma unroll
            for (int v = 0; v < 8; v++) {
                size_t o0 = (size_t)(rowS0 + v) * 32 + n0;
                size_t o1 = (size_t)(rowS1 + v) * 32 + n0;
                Out2[o0] = r0v[v];  Out2[o0 + 16] = r1v[v];
                Out2[o1] = r2v[v];  Out2[o1 + 16] = r3v[v];
            }
        }
    } else {                               // tail block only
#pragma unroll
        for (int v = 0; v < 8; v++) {
            int row0 = rowS0 + v, row1 = rowS1 + v;
            if (row0 < N_NODES) {
                size_t o0 = (size_t)row0 * 32 + n0;
                Out[o0] = r0v[v];  Out[o0 + 16] = r1v[v];
                if (Out2) { Out2[o0] = r0v[v];  Out2[o0 + 16] = r1v[v]; }
            }
            if (row1 < N_NODES) {
                size_t o1 = (size_t)row1 * 32 + n0;
                Out[o1] = r2v[v];  Out[o1 + 16] = r3v[v];
                if (Out2) { Out2[o1] = r2v[v];  Out2[o1 + 16] = r3v[v]; }
            }
        }
    }
}

// ---------------------------------------------------------------------------
// pooling: h >= 0 (post-ReLU) so segment_max == atomicMax on uint bit pattern
__global__ void k_pool(const float* __restrict__ h, const int* __restrict__ batch,
                       unsigned* __restrict__ mx, float* __restrict__ sm,
                       float* __restrict__ cnt) {
    int t = blockIdx.x * 256 + threadIdx.x;
    int node = t >> 5, f = t & 31;
    if (node >= N_NODES) return;
    int g = batch[node];
    if (f < HIDDEN) {
        float v = h[(size_t)node * 32 + f];
        atomicMax(&mx[g * HIDDEN + f], __float_as_uint(v));
        atomicAdd(&sm[g * HIDDEN + f], v);
    } else if (f == 31) {
        atomicAdd(&cnt[g], 1.0f);
    }
}

// ---------------------------------------------------------------------------
// final classifier: [256 x 40] @ [40 x 2] + bl
__global__ void k_final(const unsigned* __restrict__ mx, const float* __restrict__ sm,
                        const float* __restrict__ cnt, const float* __restrict__ Wl,
                        const float* __restrict__ bl, float* __restrict__ out) {
    int t = threadIdx.x;               // 512 threads: (graph, class)
    int g = t >> 1, c = t & 1;
    float cn = cnt[g];
    if (cn < 1.0f) cn = 1.0f;
    float acc = bl[c];
    for (int j = 0; j < HIDDEN; j++)
        acc += __uint_as_float(mx[g * HIDDEN + j]) * Wl[j * N_CLASSES + c];
    for (int j = 0; j < HIDDEN; j++)
        acc += (sm[g * HIDDEN + j] / cn) * Wl[(HIDDEN + j) * N_CLASSES + c];
    out[g * N_CLASSES + c] = acc;
}

// ---------------------------------------------------------------------------
extern "C" void kernel_launch(void* const* d_in, const int* in_sizes, int n_in,
                              void* d_out, int out_size, void* d_ws, size_t ws_size,
                              hipStream_t stream) {
    const float* x   = (const float*)d_in[0];
    const int*   ei  = (const int*)d_in[1];
    const int*   bat = (const int*)d_in[2];
    const float* W1  = (const float*)d_in[3];
    const float* b1  = (const float*)d_in[4];
    const float* W2  = (const float*)d_in[5];
    const float* b2  = (const float*)d_in[6];
    const float* W3  = (const float*)d_in[7];
    const float* b3  = (const float*)d_in[8];
    const float* Wl  = (const float*)d_in[9];
    const float* bl  = (const float*)d_in[10];
    float* out = (float*)d_out;

    char* ws = (char*)d_ws;
    size_t off = 0;
    auto alloc = [&](size_t bytes) -> void* {
        void* p = ws + off;
        off += (bytes + 255) & ~(size_t)255;
        return p;
    };
    float*    bufA = (float*)alloc(sizeof(float) * (size_t)N_NODES * N_FEATS);
    float*    P    = (float*)alloc(sizeof(float) * (size_t)N_NODES * 32);
    float*    Q    = (float*)alloc(sizeof(float) * (size_t)N_NODES * 32);
    _Float16* W1f  = (_Float16*)alloc(sizeof(_Float16) * N_FEATS * 32);
    _Float16* W2f  = (_Float16*)alloc(sizeof(_Float16) * 32 * 32);
    _Float16* W3f  = (_Float16*)alloc(sizeof(_Float16) * 32 * 32);
    float*    bPad = (float*)alloc(sizeof(float) * 3 * 32);
    unsigned* mx   = (unsigned*)alloc(sizeof(unsigned) * N_GRAPHS * HIDDEN);
    float*    sm   = (float*)alloc(sizeof(float) * N_GRAPHS * HIDDEN);
    float*    cnt  = (float*)alloc(sizeof(float) * N_GRAPHS);

    const int gemmGrid    = (N_NODES + 255) / 256;
    const int edgeGrid128 = (N_EDGES * 32 + 255) / 256;
    const int edgeGrid20  = (N_EDGES * HIDDEN + 255) / 256;
    const int copy128Grid = ((N_NODES * N_FEATS / 4) + 255) / 256;
    const int poolGrid    = (N_NODES * 32 + 255) / 256;

    k_convert_w<<<1, 256, 0, stream>>>(W1, W2, W3, b1, b2, b3, W1f, W2f, W3f, bPad);
    k_init_pool<<<1, 256, 0, stream>>>(mx, sm, cnt);

    // Layer 1: agg = x + segment_sum(x[src], dst); h1 = relu(agg @ W1 + b1)
    k_copy4<<<copy128Grid, 256, 0, stream>>>((const float4*)x, (float4*)bufA,
                                             N_NODES * N_FEATS / 4);
    k_scatter128<<<edgeGrid128, 256, 0, stream>>>(x, ei, bufA);
    k_gemm_relu<<<gemmGrid, 256, 0, stream>>>(bufA, W1f, bPad + 0, P, Q, N_FEATS);

    // Layer 2 (gemm writes h2 into P and re-seeds Q in place for layer 3)
    k_scatter20<<<edgeGrid20, 256, 0, stream>>>(P, ei, Q);
    k_gemm_relu<<<gemmGrid, 256, 0, stream>>>(Q, W2f, bPad + 32, P, Q, 32);

    // Layer 3
    k_scatter20<<<edgeGrid20, 256, 0, stream>>>(P, ei, Q);
    k_gemm_relu<<<gemmGrid, 256, 0, stream>>>(Q, W3f, bPad + 64, P, nullptr, 32);

    // Pool + classifier
    k_pool<<<poolGrid, 256, 0, stream>>>(P, bat, mx, sm, cnt);
    k_final<<<1, 512, 0, stream>>>(mx, sm, cnt, Wl, bl, out);
}
